// InferenceModel_721554506441
// MI455X (gfx1250) — matrix operand order repaired and checked by reading, not compiled, not run on hardware
//
#include <hip/hip_runtime.h>
#include <hip/hip_bf16.h>

// ---------------- problem constants ----------------
#define NNODES 50000
#define NEDGES 1600000
#define NREL   8
#define DIN    128
#define DHID   128
#define DOUT   64
#define NPOOL  256

typedef __attribute__((ext_vector_type(16))) __bf16 v16bf;
typedef __attribute__((ext_vector_type(8)))  float  v8f;
typedef __attribute__((ext_vector_type(4)))  float  v4f;

// ---------------- zero-fill (graph-capture safe, no memset) ----------------
__global__ void zero_f32v4(float* __restrict__ p, long long n4) {
    long long i = (long long)blockIdx.x * blockDim.x + threadIdx.x;
    if (i < n4) ((v4f*)p)[i] = (v4f){0.f, 0.f, 0.f, 0.f};
}

// ---------------- weight prep: f32 [R][D][H] + root [D][H] -> bf16 [(R+1)][H][D]
__global__ void prep_weights(const float* __restrict__ W, const float* __restrict__ root,
                             __bf16* __restrict__ out, int D, int H) {
    int idx = blockIdx.x * blockDim.x + threadIdx.x;
    int total = (NREL + 1) * H * D;
    if (idx >= total) return;
    int r   = idx / (H * D);
    int rem = idx % (H * D);
    int n   = rem / D;
    int k   = rem % D;
    float v = (r < NREL) ? W[((size_t)r * D + k) * H + n] : root[(size_t)k * H + n];
    out[idx] = (__bf16)v;   // out layout [r][n][k] == linear idx
}

// ---------------- edge scatter: one wave per edge, 128 f32 feature atomically added
__global__ void edge_scatter(const int* __restrict__ ei, const int* __restrict__ et,
                             const float* __restrict__ feat,
                             float* __restrict__ agg, float* __restrict__ cnt, int do_cnt) {
    int lane = threadIdx.x & 31;
    int wave = threadIdx.x >> 5;
    long long e = (long long)blockIdx.x * (blockDim.x >> 5) + wave;
    if (e >= NEDGES) return;
    int src = ei[e];
    int dst = ei[(size_t)NEDGES + e];
    int t   = et[e];
    const float* s = feat + (size_t)src * DIN + lane * 4;
    float*       d = agg  + ((size_t)t * NNODES + dst) * DIN + lane * 4;
    v4f v = *(const v4f*)s;
    unsafeAtomicAdd(d + 0, v[0]);
    unsafeAtomicAdd(d + 1, v[1]);
    unsafeAtomicAdd(d + 2, v[2]);
    unsafeAtomicAdd(d + 3, v[3]);
    if (do_cnt && lane == 0)
        unsafeAtomicAdd(&cnt[(size_t)t * NNODES + dst], 1.0f);
}

// ---------------- fused mean-normalize + (R+1)-way GEMM via WMMA bf16 ----------------
// out[n][h] = sum_r (agg[r][n][:]/max(cnt,1)) @ W_r  +  feat[n][:] @ root  + bias
// Wt: bf16 [(R+1)][H][128] (column-major per slab; slab R is root). K = 128 fixed.
template <int NT>   // NT = H/16 column tiles kept in registers
__global__ void rgcn_gemm(const float* __restrict__ agg,
                          const float* __restrict__ cnt,
                          const float* __restrict__ feat,
                          const __bf16* __restrict__ Wt,
                          const float* __restrict__ bias,
                          float* __restrict__ out, int relu) {
    constexpr int H = NT * 16;
    const int lane = threadIdx.x & 31;
    const int wave = threadIdx.x >> 5;
    const int tile = blockIdx.x * (blockDim.x >> 5) + wave;
    const int n0   = tile * 16;
    if (n0 >= NNODES) return;          // wave-uniform: EXEC all-ones inside

    const int m    = lane & 15;        // A row / D column
    const int half = lane >> 4;
    const int node = n0 + m;

    v8f acc[NT];
#pragma unroll
    for (int t = 0; t < NT; ++t) acc[t] = (v8f){0.f,0.f,0.f,0.f,0.f,0.f,0.f,0.f};

    for (int r = 0; r <= NREL; ++r) {
        const float* src = (r < NREL) ? (agg + ((size_t)r * NNODES + node) * 128)
                                      : (feat + (size_t)node * 128);
        float invc = 1.0f;
        if (r < NREL) {
            float c = cnt[(size_t)r * NNODES + node];
            invc = 1.0f / fmaxf(c, 1.0f);
        }
#pragma unroll
        for (int kc = 0; kc < 4; ++kc) {   // K = 128 in chunks of 32
            const float* p0 = src + kc * 32 + half * 16;       // K run [half*16, +8)
            const float* p1 = src + kc * 32 + half * 16 + 8;   // K run [half*16+8, +8)
            v4f a0 = *(const v4f*)(p0);
            v4f a1 = *(const v4f*)(p0 + 4);
            v4f a2 = *(const v4f*)(p1);
            v4f a3 = *(const v4f*)(p1 + 4);
            v16bf A;
#pragma unroll
            for (int i = 0; i < 4; ++i) {
                A[i]      = (__bf16)(a0[i] * invc);
                A[4 + i]  = (__bf16)(a1[i] * invc);
                A[8 + i]  = (__bf16)(a2[i] * invc);
                A[12 + i] = (__bf16)(a3[i] * invc);
            }
#pragma unroll
            for (int nt = 0; nt < NT; ++nt) {
                // B: column (nt*16 + m), contiguous K [kc*32 + half*16, +16)
                const __bf16* bp = Wt + (((size_t)r * H + nt * 16 + m) * 128)
                                      + kc * 32 + half * 16;
                v16bf B = *(const v16bf*)bp;
                acc[nt] = __builtin_amdgcn_wmma_f32_16x16x32_bf16(
                    false, A, false, B, (short)0, acc[nt], false, false);
            }
        }
    }

    // epilogue: bias (+ relu), D layout: VGPR j -> row j + 8*half, col = m
#pragma unroll
    for (int nt = 0; nt < NT; ++nt) {
        int col  = nt * 16 + m;
        float bv = bias[col];
#pragma unroll
        for (int j = 0; j < 8; ++j) {
            int row = n0 + j + half * 8;
            float v = acc[nt][j] + bv;
            if (relu) v = fmaxf(v, 0.f);
            out[(size_t)row * H + col] = v;
        }
    }
}

// ---------------- weighted pooling over 256 selected nodes ----------------
__global__ void pool_kernel(const float* __restrict__ x, const int* __restrict__ pidx,
                            const float* __restrict__ emb, float* __restrict__ out) {
    __shared__ float wv[NPOOL];
    __shared__ float swsh;
    int tid = threadIdx.x;
    int p = pidx[tid];
    wv[tid] = 4.0f * x[(size_t)p * DIN + 0] + 1.0f * x[(size_t)p * DIN + 1]
            + 2.0f * x[(size_t)p * DIN + 2];
    __syncthreads();
    if (tid == 0) {
        float s = 0.f;
        for (int i = 0; i < NPOOL; ++i) s += wv[i];
        swsh = s + 1e-9f;
    }
    __syncthreads();
    if (tid < DOUT) {
        float s = 0.f;
        for (int i = 0; i < NPOOL; ++i)
            s += emb[(size_t)pidx[i] * DOUT + tid] * wv[i];
        out[tid] = s / swsh;
    }
}

// ---------------- launch ----------------
extern "C" void kernel_launch(void* const* d_in, const int* in_sizes, int n_in,
                              void* d_out, int out_size, void* d_ws, size_t ws_size,
                              hipStream_t stream) {
    const float* x     = (const float*)d_in[0];
    const int*   ei    = (const int*)d_in[1];
    const int*   et    = (const int*)d_in[2];
    const int*   pidx  = (const int*)d_in[3];
    const float* W1    = (const float*)d_in[4];
    const float* root1 = (const float*)d_in[5];
    const float* b1    = (const float*)d_in[6];
    const float* W2    = (const float*)d_in[7];
    const float* root2 = (const float*)d_in[8];
    const float* b2    = (const float*)d_in[9];
    float* outp = (float*)d_out;

    // workspace layout (bytes, 256-aligned offsets)
    char* ws = (char*)d_ws;
    const size_t AGG_B = (size_t)NREL * NNODES * 128 * 4;   // 204.8 MB (reused by both layers)
    const size_t CNT_B = (size_t)NREL * NNODES * 4;         // 1.6 MB
    const size_t H_B   = (size_t)NNODES * DHID * 4;         // 25.6 MB
    const size_t E_B   = (size_t)NNODES * DOUT * 4;         // 12.8 MB
    const size_t W1_B  = (size_t)(NREL + 1) * DHID * 128 * 2;
    float*  agg  = (float*)(ws);
    float*  cnt  = (float*)(ws + AGG_B);
    float*  hbuf = (float*)(ws + AGG_B + CNT_B);
    float*  emb  = (float*)(ws + AGG_B + CNT_B + H_B);
    __bf16* Wt1  = (__bf16*)(ws + AGG_B + CNT_B + H_B + E_B);
    __bf16* Wt2  = (__bf16*)(ws + AGG_B + CNT_B + H_B + E_B + W1_B);

    const long long agg4 = (long long)NREL * NNODES * 128 / 4;
    const long long cnt4 = (long long)NREL * NNODES / 4;

    // zero accumulators + counts
    zero_f32v4<<<(unsigned)((agg4 + 255) / 256), 256, 0, stream>>>(agg, agg4);
    zero_f32v4<<<(unsigned)((cnt4 + 255) / 256), 256, 0, stream>>>(cnt, cnt4);

    // transpose/convert weights (root folded as relation 8)
    prep_weights<<<((NREL + 1) * DHID * 128 + 255) / 256, 256, 0, stream>>>(W1, root1, Wt1, 128, DHID);
    prep_weights<<<((NREL + 1) * DOUT * 128 + 255) / 256, 256, 0, stream>>>(W2, root2, Wt2, 128, DOUT);

    // ---- layer 1 ----
    edge_scatter<<<NEDGES / 8, 256, 0, stream>>>(ei, et, x, agg, cnt, 1);
    {
        int tiles = NNODES / 16;                       // 3125
        rgcn_gemm<DHID / 16><<<(tiles + 7) / 8, 256, 0, stream>>>(agg, cnt, x, Wt1, b1, hbuf, 1);
    }

    // ---- layer 2 (reuse agg; counts unchanged) ----
    zero_f32v4<<<(unsigned)((agg4 + 255) / 256), 256, 0, stream>>>(agg, agg4);
    edge_scatter<<<NEDGES / 8, 256, 0, stream>>>(ei, et, hbuf, agg, cnt, 0);
    {
        int tiles = NNODES / 16;
        rgcn_gemm<DOUT / 16><<<(tiles + 7) / 8, 256, 0, stream>>>(agg, cnt, hbuf, Wt2, b2, emb, 0);
    }

    // ---- pooling ----
    pool_kernel<<<1, NPOOL, 0, stream>>>(x, pidx, emb, outp);
}
